// DGCNNv2Seg_57664230916164
// MI455X (gfx1250) — matrix operand order repaired and checked by reading, not compiled, not run on hardware
//
#include <hip/hip_runtime.h>
#include <hip/hip_bf16.h>

#define NPTS 4096
#define BATCH 4
#define KNN 20

typedef __attribute__((ext_vector_type(16))) _Float16 v16h;
typedef __attribute__((ext_vector_type(8)))  float    v8f;

// ---------------------------------------------------------------------------
// helpers
// ---------------------------------------------------------------------------
__device__ __forceinline__ float warp_sum16(float v) {
  // reduce across the 16 lanes of each half-wave (lane bits 0..3)
  v += __shfl_xor(v, 1, 32);
  v += __shfl_xor(v, 2, 32);
  v += __shfl_xor(v, 4, 32);
  v += __shfl_xor(v, 8, 32);
  return v;
}

__device__ __forceinline__ int a_k_of_e(int e, int half) {
  // contiguous k order: k = 16*half + e (matches B fill)
  int j = e >> 1, r = e & 1;
  return 16 * half + 2 * j + r;
}

// ---------------------------------------------------------------------------
// small elementwise kernels
// ---------------------------------------------------------------------------
__global__ void zero_kernel(float* p, int n) {
  int i = blockIdx.x * 256 + threadIdx.x;
  if (i < n) p[i] = 0.f;
}

__global__ void transpose_pts(const float* __restrict__ pts, float* __restrict__ x0) {
  int i = blockIdx.x * 256 + threadIdx.x;           // over B*N
  int b = i / NPTS, n = i % NPTS;
  #pragma unroll
  for (int c = 0; c < 3; ++c)
    x0[((size_t)b * 3 + c) * NPTS + n] = pts[((size_t)b * NPTS + n) * 3 + c];
}

__global__ void sumsq_kernel(const float* __restrict__ x, int stride, int off, int C,
                             float* __restrict__ xx) {
  int i = blockIdx.x * 256 + threadIdx.x;           // over B*N
  int b = i / NPTS, n = i % NPTS;
  const float* xb = x + ((size_t)b * stride + off) * NPTS + n;
  float s = 0.f;
  for (int c = 0; c < C; ++c) { float v = xb[(size_t)c * NPTS]; s += v * v; }
  xx[i] = s;
}

__global__ void finalize_stats(const float* __restrict__ ssum, const float* __restrict__ ssq,
                               const float* __restrict__ gamma, const float* __restrict__ beta,
                               float invcount, int O,
                               float* __restrict__ scale, float* __restrict__ shift) {
  int o = blockIdx.x * 64 + threadIdx.x;
  if (o < O) {
    float m = ssum[o] * invcount;
    float v = ssq[o] * invcount - m * m;
    float s = gamma[o] * rsqrtf(v + 1e-5f);
    scale[o] = s;
    shift[o] = beta[o] - s * m;
  }
}

__global__ void bn_lrelu_apply(float* __restrict__ y, const float* __restrict__ scale,
                               const float* __restrict__ shift, int O) {
  int i = blockIdx.x * 256 + threadIdx.x;           // over B*O*N
  int o = (i / NPTS) % O;
  float v = y[i] * scale[o] + shift[o];
  y[i] = (v >= 0.f) ? v : 0.2f * v;
}

__global__ __launch_bounds__(256) void colmax_kernel(const float* __restrict__ x5,
                                                     float* __restrict__ gmax, int O) {
  int o = blockIdx.x, b = blockIdx.y;
  const float* p = x5 + ((size_t)b * O + o) * NPTS;
  float mv = -3.4e38f;
  for (int i = threadIdx.x; i < NPTS; i += 256) mv = fmaxf(mv, p[i]);
  __shared__ float red[256];
  red[threadIdx.x] = mv;
  __syncthreads();
  for (int s = 128; s > 0; s >>= 1) {
    if (threadIdx.x < s) red[threadIdx.x] = fmaxf(red[threadIdx.x], red[threadIdx.x + s]);
    __syncthreads();
  }
  if (threadIdx.x == 0) gmax[b * O + o] = red[0];
}

// gterm[b,o] = sum_c Ws1[o, 512+c] * gmax[b,c]
__global__ void gterm_kernel(const float* __restrict__ Ws1, const float* __restrict__ gmax,
                             float* __restrict__ gterm) {
  int o = threadIdx.x;                              // 256 threads
  int b = blockIdx.y;
  float s = 0.f;
  for (int c = 0; c < 512; ++c) s += Ws1[(size_t)o * 1024 + 512 + c] * gmax[b * 512 + c];
  gterm[b * 256 + o] = s;
}

// out[b,{0,1},n] = Ws3 @ h2 + bias3
__global__ void final_kernel(const float* __restrict__ Ws3, const float* __restrict__ bias3,
                             const float* __restrict__ h2, float* __restrict__ out) {
  int i = blockIdx.x * 256 + threadIdx.x;           // over B*N
  int b = i / NPTS, n = i % NPTS;
  const float* hb = h2 + (size_t)b * 128 * NPTS + n;
  float s0 = bias3[0], s1 = bias3[1];
  for (int c = 0; c < 128; ++c) {
    float hv = hb[(size_t)c * NPTS];
    s0 += Ws3[c] * hv;
    s1 += Ws3[128 + c] * hv;
  }
  out[((size_t)b * 2 + 0) * NPTS + n] = s0;
  out[((size_t)b * 2 + 1) * NPTS + n] = s1;
}

// ---------------------------------------------------------------------------
// kNN top-20 (includes self, matching top_k of neg_dist). f32 exact ranking.
// One 64-thread workgroup per query point.
// ---------------------------------------------------------------------------
__global__ __launch_bounds__(64)
void knn_kernel(const float* __restrict__ x, int stride, int off, int C,
                const float* __restrict__ xx, int* __restrict__ idxout) {
  int n = blockIdx.x, b = blockIdx.y;
  int t = threadIdx.x;
  __shared__ float centerLds[128];
  __shared__ float cval[64 * KNN];
  __shared__ int   cidx[64 * KNN];
  __shared__ float rv[64];
  __shared__ int   hS[64];

  const float* xb = x + ((size_t)b * stride + off) * NPTS;
  for (int c = t; c < C; c += 64) centerLds[c] = xb[(size_t)c * NPTS + n];
  __syncthreads();

  float bv[KNN]; int bi[KNN];
  #pragma unroll
  for (int j = 0; j < KNN; ++j) { bv[j] = -3.4e38f; bi[j] = 0; }

  const float* xxb = xx + (size_t)b * NPTS;
  for (int m = t; m < NPTS; m += 64) {
    float inner = 0.f;
    for (int c = 0; c < C; ++c) inner += centerLds[c] * xb[(size_t)c * NPTS + m];
    float score = 2.f * inner - xxb[m];   // xx[n] shift is rank-invariant
    if (score > bv[KNN - 1]) {
      float v = score; int vi = m;
      #pragma unroll
      for (int j = 0; j < KNN; ++j) {
        if (v > bv[j]) { float tv = bv[j]; int ti = bi[j]; bv[j] = v; bi[j] = vi; v = tv; vi = ti; }
      }
    }
  }
  #pragma unroll
  for (int j = 0; j < KNN; ++j) { cval[t * KNN + j] = bv[j]; cidx[t * KNN + j] = bi[j]; }
  hS[t] = 0;
  __syncthreads();

  int* od = idxout + ((size_t)b * NPTS + n) * KNN;
  for (int round = 0; round < KNN; ++round) {
    int h = hS[t];
    rv[t] = (h < KNN) ? cval[t * KNN + h] : -3.4e38f;
    __syncthreads();
    if (t == 0) {
      int tw = 0; float best = rv[0];
      for (int q = 1; q < 64; ++q) if (rv[q] > best) { best = rv[q]; tw = q; }
      int hw = hS[tw];
      od[round] = cidx[tw * KNN + hw];
      hS[tw] = hw + 1;
    }
    __syncthreads();
  }
}

// ---------------------------------------------------------------------------
// Fused EdgeConv GEMM via WMMA. Workgroup = 4 points x 20 nbrs = 80 columns.
// PASS 0: accumulate BN stats.  PASS 1: BN+lrelu, max over k, write output.
// ---------------------------------------------------------------------------
template <int PASS>
__global__ __launch_bounds__(256)
void edge_conv_kernel(const float* __restrict__ Wg,                     // (O x 2C) row-major
                      const float* __restrict__ xin, int instride, int inoff, int C,
                      const int* __restrict__ idxb, int O,
                      float* stats_sum, float* stats_sumsq,             // PASS 0
                      const float* __restrict__ scale, const float* __restrict__ shift,
                      float* __restrict__ outp, int outstride, int outoff) { // PASS 1
  const int K2 = 2 * C;
  const int KP = (K2 + 31) & ~31;
  const int b = blockIdx.y;
  const int n0 = blockIdx.x * 4;
  const int tid = threadIdx.x;
  const int lane = tid & 31;
  const int wave = tid >> 5;
  const int Ml = lane & 15;
  const int half = lane >> 4;

  __shared__ int nbr[80];
  __shared__ _Float16 fLds[256 * 80];   // 40 KB
  __shared__ float yLds[32 * 80];       // 10 KB (PASS 1)

  if (tid < 80) {
    int p = tid / KNN, kk = tid % KNN;
    nbr[tid] = idxb[((size_t)b * NPTS + (n0 + p)) * KNN + kk];
  }
  __syncthreads();

  const float* xb = xin + ((size_t)b * instride + inoff) * NPTS;
  for (int i = tid; i < C * 80; i += 256) {
    int c = i / 80, col = i % 80;
    int nn = n0 + col / KNN;
    int g = nbr[col];
    float xc = xb[(size_t)c * NPTS + nn];
    float xg = xb[(size_t)c * NPTS + g];
    fLds[c * 80 + col] = (_Float16)(xg - xc);
    fLds[(C + c) * 80 + col] = (_Float16)xc;
  }
  for (int i = K2 * 80 + tid; i < KP * 80; i += 256) fLds[i] = (_Float16)0.f;
  __syncthreads();

  const int nKs = KP >> 5;
  const int nMC = O >> 5;  // O is a multiple of 32

  for (int mc = 0; mc < nMC; ++mc) {
    for (int t = wave; t < 10; t += 8) {     // 2 m-tiles x 5 col-tiles
      int mt = t / 5, ct = t % 5;
      int m_base = mc * 32 + mt * 16;
      int col = ct * 16 + Ml;
      v8f acc = {};
      for (int ks = 0; ks < nKs; ++ks) {
        v16h a, bf;
        #pragma unroll
        for (int e = 0; e < 16; ++e) {
          int kidx = ks * 32 + a_k_of_e(e, half);
          float wv = (kidx < K2) ? Wg[(size_t)(m_base + Ml) * K2 + kidx] : 0.f;
          a[e] = (_Float16)wv;
        }
        #pragma unroll
        for (int e = 0; e < 16; ++e) {
          int krow = ks * 32 + 16 * half + e;
          bf[e] = fLds[krow * 80 + col];
        }
        acc = __builtin_amdgcn_wmma_f32_16x16x32_f16(false, a, false, bf, (short)0, acc,
                                                     false, false);
      }
      if (PASS == 0) {
        #pragma unroll
        for (int r = 0; r < 8; ++r) {
          float v = acc[r];
          float s = warp_sum16(v);
          float s2 = warp_sum16(v * v);
          if (Ml == 0) {
            int o = m_base + 8 * half + r;
            atomicAdd(&stats_sum[o], s);
            atomicAdd(&stats_sumsq[o], s2);
          }
        }
      } else {
        #pragma unroll
        for (int r = 0; r < 8; ++r) {
          int mloc = mt * 16 + 8 * half + r;
          yLds[mloc * 80 + col] = acc[r];
        }
      }
    }
    if (PASS == 1) {
      __syncthreads();
      if (tid < 128) {                      // 32 rows x 4 points
        int p = tid & 3;
        int rloc = tid >> 2;
        int o = mc * 32 + rloc;
        float a_s = scale[o], c_s = shift[o];
        float mv = -3.4e38f;
        #pragma unroll
        for (int kk = 0; kk < KNN; ++kk) {
          float v = yLds[rloc * 80 + p * KNN + kk];
          v = a_s * v + c_s;
          v = (v >= 0.f) ? v : 0.2f * v;
          mv = fmaxf(mv, v);
        }
        outp[((size_t)b * outstride + outoff + o) * NPTS + (n0 + p)] = mv;
      }
      __syncthreads();
    }
  }
}

// ---------------------------------------------------------------------------
// 1x1 conv GEMM via WMMA: Y[b,o,col] = sum_k W[o,k] * X[b,k,col] (+ addvec[b,o])
// Also accumulates per-channel BN stats. Workgroup owns a 64x64 output tile.
// ---------------------------------------------------------------------------
__global__ __launch_bounds__(256)
void gemm_kernel(const float* __restrict__ Wg, int wstride,
                 const float* __restrict__ X, int xstride, int xoff, int K2, int O,
                 const float* __restrict__ addvec,
                 float* __restrict__ Y,
                 float* stats_sum, float* stats_sumsq) {
  int b = blockIdx.z;
  int colbase = blockIdx.x * 64;
  int mbase64 = blockIdx.y * 64;
  int tid = threadIdx.x, lane = tid & 31, wave = tid >> 5;
  int Ml = lane & 15, half = lane >> 4;
  const float* xb = X + ((size_t)b * xstride + xoff) * NPTS;

  v8f zero8 = {};
  v8f acc[2]; acc[0] = zero8; acc[1] = zero8;

  for (int ks = 0; ks < K2 / 32; ++ks) {
    #pragma unroll
    for (int u = 0; u < 2; ++u) {
      int t = wave + u * 8;
      int mt = t >> 2, ct = t & 3;
      int m = mbase64 + mt * 16 + Ml;
      int col = colbase + ct * 16 + Ml;
      v16h a, bf;
      #pragma unroll
      for (int e = 0; e < 16; ++e)
        a[e] = (_Float16)Wg[(size_t)m * wstride + ks * 32 + a_k_of_e(e, half)];
      #pragma unroll
      for (int e = 0; e < 16; ++e)
        bf[e] = (_Float16)xb[(size_t)(ks * 32 + 16 * half + e) * NPTS + col];
      acc[u] = __builtin_amdgcn_wmma_f32_16x16x32_f16(false, a, false, bf, (short)0, acc[u],
                                                      false, false);
    }
  }

  #pragma unroll
  for (int u = 0; u < 2; ++u) {
    int t = wave + u * 8;
    int mt = t >> 2, ct = t & 3;
    int col = colbase + ct * 16 + Ml;
    #pragma unroll
    for (int r = 0; r < 8; ++r) {
      int o = mbase64 + mt * 16 + 8 * half + r;
      float v = acc[u][r];
      if (addvec) v += addvec[b * O + o];
      Y[((size_t)b * O + o) * NPTS + col] = v;
      float s = warp_sum16(v);
      float s2 = warp_sum16(v * v);
      if (Ml == 0) {
        atomicAdd(&stats_sum[o], s);
        atomicAdd(&stats_sumsq[o], s2);
      }
    }
  }
}

// ---------------------------------------------------------------------------
// host orchestration
// ---------------------------------------------------------------------------
extern "C" void kernel_launch(void* const* d_in, const int* in_sizes, int n_in,
                              void* d_out, int out_size, void* d_ws, size_t ws_size,
                              hipStream_t stream) {
  (void)in_sizes; (void)n_in; (void)out_size; (void)ws_size;
  const float* pts = (const float*)d_in[0];
  const float *W1 = (const float*)d_in[1],  *g1 = (const float*)d_in[2],  *b1 = (const float*)d_in[3];
  const float *W2 = (const float*)d_in[4],  *g2 = (const float*)d_in[5],  *b2 = (const float*)d_in[6];
  const float *W3 = (const float*)d_in[7],  *g3 = (const float*)d_in[8],  *b3 = (const float*)d_in[9];
  const float *W4 = (const float*)d_in[10], *g4 = (const float*)d_in[11], *b4 = (const float*)d_in[12];
  const float *W5 = (const float*)d_in[13], *g5 = (const float*)d_in[14], *b5 = (const float*)d_in[15];
  const float *Ws1 = (const float*)d_in[16], *gs1 = (const float*)d_in[17], *bs1 = (const float*)d_in[18];
  const float *Ws2 = (const float*)d_in[19], *gs2 = (const float*)d_in[20], *bs2 = (const float*)d_in[21];
  const float *Ws3 = (const float*)d_in[22], *bias3 = (const float*)d_in[23];

  float* ws = (float*)d_ws;
  size_t o = 0;
  float* x0    = ws + o; o += (size_t)BATCH * 3 * NPTS;
  float* xc    = ws + o; o += (size_t)BATCH * 512 * NPTS;   // x1..x4 slices
  float* x5    = ws + o; o += (size_t)BATCH * 512 * NPTS;
  float* h1    = ws + o; o += (size_t)BATCH * 256 * NPTS;
  float* h2    = ws + o; o += (size_t)BATCH * 128 * NPTS;
  float* xxb   = ws + o; o += (size_t)BATCH * NPTS;
  float* gmax  = ws + o; o += (size_t)BATCH * 512;
  float* gterm = ws + o; o += (size_t)BATCH * 256;
  float* ssum  = ws + o; o += 512;
  float* ssq   = ws + o; o += 512;
  float* scale = ws + o; o += 512;
  float* shift = ws + o; o += 512;
  int*   idx   = (int*)(ws + o);

  const int BN = BATCH * NPTS;

  auto edge_stage = [&](const float* Wg, const float* gam, const float* bet,
                        const float* xin, int instride, int inoff, int C, int O,
                        float* outb, int outstride, int outoff) {
    sumsq_kernel<<<BN / 256, 256, 0, stream>>>(xin, instride, inoff, C, xxb);
    knn_kernel<<<dim3(NPTS, BATCH), 64, 0, stream>>>(xin, instride, inoff, C, xxb, idx);
    zero_kernel<<<4, 256, 0, stream>>>(ssum, 1024);  // ssum + ssq contiguous
    edge_conv_kernel<0><<<dim3(NPTS / 4, BATCH), 256, 0, stream>>>(
        Wg, xin, instride, inoff, C, idx, O, ssum, ssq, nullptr, nullptr, nullptr, 0, 0);
    finalize_stats<<<(O + 63) / 64, 64, 0, stream>>>(ssum, ssq, gam, bet,
                                                     1.0f / (float)(BN * KNN), O, scale, shift);
    edge_conv_kernel<1><<<dim3(NPTS / 4, BATCH), 256, 0, stream>>>(
        Wg, xin, instride, inoff, C, idx, O, nullptr, nullptr, scale, shift,
        outb, outstride, outoff);
  };

  auto conv_stage = [&](const float* Wg, int wstride, const float* X, int xstride, int xoff,
                        int K2, int O, const float* addvec, float* Y,
                        const float* gam, const float* bet) {
    zero_kernel<<<4, 256, 0, stream>>>(ssum, 1024);
    gemm_kernel<<<dim3(NPTS / 64, O / 64, BATCH), 256, 0, stream>>>(
        Wg, wstride, X, xstride, xoff, K2, O, addvec, Y, ssum, ssq);
    finalize_stats<<<(O + 63) / 64, 64, 0, stream>>>(ssum, ssq, gam, bet, 1.0f / (float)BN,
                                                     O, scale, shift);
    bn_lrelu_apply<<<(BATCH * O * NPTS) / 256, 256, 0, stream>>>(Y, scale, shift, O);
  };

  transpose_pts<<<BN / 256, 256, 0, stream>>>(pts, x0);

  edge_stage(W1, g1, b1, x0, 3, 0, 3, 64, xc, 512, 0);       // x1 -> xc[0:64]
  edge_stage(W2, g2, b2, xc, 512, 0, 64, 64, xc, 512, 64);   // x2 -> xc[64:128]
  edge_stage(W3, g3, b3, xc, 512, 64, 64, 128, xc, 512, 128);// x3 -> xc[128:256]
  edge_stage(W4, g4, b4, xc, 512, 128, 128, 256, xc, 512, 256);// x4 -> xc[256:512]

  conv_stage(W5, 512, xc, 512, 0, 512, 512, nullptr, x5, g5, b5);           // x5
  colmax_kernel<<<dim3(512, BATCH), 256, 0, stream>>>(x5, gmax, 512);       // g
  gterm_kernel<<<dim3(1, BATCH), 256, 0, stream>>>(Ws1, gmax, gterm);       // Ws1[:,512:] @ g
  conv_stage(Ws1, 1024, xc, 512, 0, 512, 256, gterm, h1, gs1, bs1);         // h1
  conv_stage(Ws2, 256, h1, 256, 0, 256, 128, nullptr, h2, gs2, bs2);        // h2

  final_kernel<<<BN / 256, 256, 0, stream>>>(Ws3, bias3, h2, (float*)d_out);
}
